// CRLEncoder_91225105367065
// MI455X (gfx1250) — matrix operand order repaired and checked by reading, not compiled, not run on hardware
//
#include <hip/hip_runtime.h>
#include <cstdint>
#include <cstddef>

typedef unsigned short u16;
typedef unsigned int   u32;

typedef __attribute__((ext_vector_type(16))) __bf16 v16bf;
typedef __attribute__((ext_vector_type(2)))  __bf16 v2bf;
typedef __attribute__((ext_vector_type(8)))  float  v8f;
typedef __attribute__((ext_vector_type(4)))  int    v4i;

struct __attribute__((aligned(16))) U4 { u32 x, y, z, w; };
struct U4x2 { U4 lo, hi; };

#define AS1 __attribute__((address_space(1)))
#define AS3 __attribute__((address_space(3)))

// ---- feature detection: gfx1250 async global->LDS copies -------------------
#if defined(__HIP_DEVICE_COMPILE__) && defined(__has_builtin)
#if __has_builtin(__builtin_amdgcn_global_load_async_to_lds_b128)
#define ASYNC_LDS 1
#endif
#endif

// ---- helpers ---------------------------------------------------------------
// Native fp32->bf16 (RNE) — lowers to the hardware cvt ops instead of the
// integer bit-twiddle sequence.
__device__ __forceinline__ u16 f2bf(float f) {
  __bf16 b = (__bf16)f;
  return __builtin_bit_cast(u16, b);
}

__device__ __forceinline__ u32 pack2bf(float x, float y) {
  v2bf p = {(__bf16)x, (__bf16)y};
  return __builtin_bit_cast(u32, p);
}

__device__ __forceinline__ void copy16_g2lds(u16* lds, const u16* g) {
#ifdef ASYNC_LDS
  __builtin_amdgcn_global_load_async_to_lds_b128(
      (AS1 v4i*)g, (AS3 v4i*)lds, /*imm offset*/0, /*cpol*/0);
#else
  *(U4*)lds = *(const U4*)g;
#endif
}

// Wait until at most WATERMARK async global->LDS copies remain outstanding.
template <int WATERMARK>
__device__ __forceinline__ void wait_async_copies() {
#ifdef ASYNC_LDS
#if __has_builtin(__builtin_amdgcn_s_wait_asynccnt)
  __builtin_amdgcn_s_wait_asynccnt(WATERMARK);
#else
  asm volatile("s_wait_asynccnt %0" ::"i"(WATERMARK) : "memory");
#endif
#endif
}

__device__ __forceinline__ v8f wmma_bf16(v16bf a, v16bf b, v8f c) {
  return __builtin_amdgcn_wmma_f32_16x16x32_bf16(
      false, a, false, b, (short)0, c, false, false);
}

__device__ __forceinline__ v16bf lds_frag(const u16* p, int hiOffHalfs) {
  U4x2 t;
  t.lo = *(const U4*)p;
  t.hi = *(const U4*)(p + hiOffHalfs);
  return __builtin_bit_cast(v16bf, t);
}

// ---- weight fp32 -> bf16 conversion ---------------------------------------
__global__ __launch_bounds__(256) void f32_to_bf16_v4(
    const float4* __restrict__ src, u32* __restrict__ dst, int n4) {
  int i = blockIdx.x * 256 + threadIdx.x;
  if (i < n4) {
    float4 f = src[i];
    dst[2 * i + 0] = pack2bf(f.x, f.y);
    dst[2 * i + 1] = pack2bf(f.z, f.w);
  }
}

// ---- GEMM: C[M,N] = A[M,K] * B[N,K]^T  (bf16 WMMA, fp32 accum) ------------
// Block tile 128x128, BK=32, 256 threads (8 waves), wave tile 32x64 (2x4 WMMA)
// Double-buffered LDS; async global->LDS copies for tile k+1 overlap the
// WMMAs of tile k (s_wait_asynccnt to a nonzero watermark, then barrier).
template <bool ABF16>
__global__ __launch_bounds__(256) void gemm_wmma_bf16(
    const void* __restrict__ Av, const u16* __restrict__ Bw,
    float* __restrict__ C, int M, int N, int K) {
  constexpr int PITCH = 40;  // halfs per LDS row (80B rows: conflict-free b128)
  // async instructions issued per wave per tile: B tile = 2 (+2 for bf16 A)
  constexpr int ASYNC_PER_TILE = ABF16 ? 4 : 2;

  __shared__ u16 As[2][128 * PITCH];
  __shared__ u16 Bs[2][128 * PITCH];

  const int tid   = threadIdx.x;
  const int lane  = tid & 31;
  const int wave  = tid >> 5;
  const int waveM = wave >> 1;          // 0..3 -> 32-row slice
  const int waveN = wave & 1;           // 0..1 -> 64-col slice
  const int mBase = blockIdx.y * 128;
  const int nBase = blockIdx.x * 128;

  const int lhalf = lane >> 4;
  const int l15   = lane & 15;
  const int aRow  = waveM * 32 + l15;   // A frag row in LDS tile
  const int aOff  = lhalf * 8;          // A: halves [k..k+7] then [k+16..k+23]
  const int bRow  = waveN * 64 + l15;   // B frag row (N index) in LDS tile
  const int bOff  = lhalf * 8;          // B: halves [k..k+7] then [k+16..k+23]

  // cooperative-copy mapping: 128 rows x 4 x 16B segments = 512 chunks
  const int cRow  = tid >> 2;           // 0..63 (also +64)
  const int cSeg  = tid & 3;
  const int fRow0 = tid >> 3;           // fp32-A path: 128 rows x 8 float4
  const int fSeg  = tid & 7;

  v8f acc[2][4];
#pragma unroll
  for (int i = 0; i < 2; ++i)
#pragma unroll
    for (int j = 0; j < 4; ++j) acc[i][j] = v8f{};

  const int KT = K >> 5;

  // ---------------- prologue: stage tile 0 into buffer 0 ----------------
  if constexpr (ABF16) {
    const u16* A = (const u16*)Av;
#pragma unroll
    for (int j = 0; j < 2; ++j) {
      int row = cRow + 64 * j;
      copy16_g2lds(&As[0][row * PITCH + cSeg * 8],
                   &A[(size_t)(mBase + row) * K + cSeg * 8]);
    }
  } else {
    const float* A = (const float*)Av;
#pragma unroll
    for (int j = 0; j < 4; ++j) {
      int row = fRow0 + 32 * j;
      float4 f = *(const float4*)&A[(size_t)(mBase + row) * K + fSeg * 4];
      u32* d = (u32*)&As[0][row * PITCH + fSeg * 4];
      d[0] = pack2bf(f.x, f.y);
      d[1] = pack2bf(f.z, f.w);
    }
  }
#pragma unroll
  for (int j = 0; j < 2; ++j) {
    int row = cRow + 64 * j;
    copy16_g2lds(&Bs[0][row * PITCH + cSeg * 8],
                 &Bw[(size_t)(nBase + row) * K + cSeg * 8]);
  }

  float4 fA[4];  // fp32-A register staging for the next tile
  for (int kt = 0; kt < KT; ++kt) {
    const bool haveNext = (kt + 1) < KT;
    const int kn = (kt + 1) * 32;
    const int nb = (kt + 1) & 1;
    const int cb = kt & 1;

    // ---- issue copies for tile kt+1, then wait for tile kt's copies ----
    if (haveNext) {
      if constexpr (ABF16) {
        const u16* A = (const u16*)Av;
#pragma unroll
        for (int j = 0; j < 2; ++j) {
          int row = cRow + 64 * j;
          copy16_g2lds(&As[nb][row * PITCH + cSeg * 8],
                       &A[(size_t)(mBase + row) * K + kn + cSeg * 8]);
        }
      } else {
        const float* A = (const float*)Av;
#pragma unroll
        for (int j = 0; j < 4; ++j) {
          int row = fRow0 + 32 * j;
          fA[j] = *(const float4*)&A[(size_t)(mBase + row) * K + kn + fSeg * 4];
        }
      }
#pragma unroll
      for (int j = 0; j < 2; ++j) {
        int row = cRow + 64 * j;
        copy16_g2lds(&Bs[nb][row * PITCH + cSeg * 8],
                     &Bw[(size_t)(nBase + row) * K + kn + cSeg * 8]);
      }
      wait_async_copies<ASYNC_PER_TILE>();  // tile kt done; kt+1 in flight
    } else {
      wait_async_copies<0>();
    }
    __syncthreads();

    // ---- compute tile kt: 2 A-frags x 4 B-frags -> 8 WMMAs ----
    v16bf a0 = lds_frag(&As[cb][(aRow)      * PITCH + aOff], 16);
    v16bf a1 = lds_frag(&As[cb][(aRow + 16) * PITCH + aOff], 16);
    v16bf b[4];
#pragma unroll
    for (int j = 0; j < 4; ++j)
      b[j] = lds_frag(&Bs[cb][(bRow + 16 * j) * PITCH + bOff], 16);
#pragma unroll
    for (int j = 0; j < 4; ++j) {
      acc[0][j] = wmma_bf16(a0, b[j], acc[0][j]);
      acc[1][j] = wmma_bf16(a1, b[j], acc[1][j]);
    }

    // ---- fp32-A path: convert staged registers and store for tile kt+1 ----
    if (haveNext) {
      if constexpr (!ABF16) {
#pragma unroll
        for (int j = 0; j < 4; ++j) {
          int row = fRow0 + 32 * j;
          u32* d = (u32*)&As[nb][row * PITCH + fSeg * 4];
          d[0] = pack2bf(fA[j].x, fA[j].y);
          d[1] = pack2bf(fA[j].z, fA[j].w);
        }
      }
    }
    __syncthreads();
  }

  // ---- write out (C/D layout: vgpr v -> row v, +8 for upper lane half) ----
  const int cm = mBase + waveM * 32;
  const int cn = nBase + waveN * 64 + l15;
#pragma unroll
  for (int v = 0; v < 8; ++v) {
    int r = v + lhalf * 8;
#pragma unroll
    for (int j = 0; j < 4; ++j) {
      C[(size_t)(cm + r)      * N + cn + 16 * j] = acc[0][j][v];
      C[(size_t)(cm + 16 + r) * N + cn + 16 * j] = acc[1][j][v];
    }
  }
}

// ---- fused 3-layer EMA scan over L, emits bf16 for the up-proj GEMM -------
__global__ __launch_bounds__(256) void ema_scan3(
    const float* __restrict__ h, const float* __restrict__ log_a,
    u16* __restrict__ hb) {
  constexpr int L = 4096, Di = 512;
  int tid = blockIdx.x * 256 + threadIdx.x;   // 0..2047
  int e = tid & (Di - 1);
  int b = tid >> 9;

  float a1 = 1.f / (1.f + __expf(-log_a[0 * Di + e]));
  float a2 = 1.f / (1.f + __expf(-log_a[1 * Di + e]));
  float a3 = 1.f / (1.f + __expf(-log_a[2 * Di + e]));
  float c1 = 1.f - a1, c2 = 1.f - a2, c3 = 1.f - a3;

  const float* p = h + (size_t)b * L * Di + e;
  u16* q = hb + (size_t)b * L * Di + e;

  float y1 = 0.f, y2 = 0.f, y3 = 0.f;
  for (int t = 0; t < L; ++t) {
    float hv = p[(size_t)t * Di];
    y1 = fmaf(a1, y1, c1 * hv);
    y2 = fmaf(a2, y2, c2 * y1);
    y3 = fmaf(a3, y3, c3 * y2);
    q[(size_t)t * Di] = f2bf(y3);
  }
}

// ---------------------------------------------------------------------------
extern "C" void kernel_launch(void* const* d_in, const int* in_sizes, int n_in,
                              void* d_out, int out_size, void* d_ws,
                              size_t ws_size, hipStream_t stream) {
  (void)in_sizes; (void)n_in; (void)out_size; (void)ws_size;
  constexpr int Bb = 4, L = 4096, D = 2048, Di = 512;
  constexpr int M = Bb * L;                      // 16384

  const float* x     = (const float*)d_in[0];    // [B,L,D]
  const float* Wd    = (const float*)d_in[1];    // [Di,D]
  const float* Wu    = (const float*)d_in[2];    // [D,Di]
  const float* log_a = (const float*)d_in[3];    // [3,Di]
  float* out = (float*)d_out;                    // [B,L,D]

  char* ws = (char*)d_ws;
  u16*   Wd_b = (u16*)(ws + 0);                  // 2 MiB
  u16*   Wu_b = (u16*)(ws + (2u << 20));         // 2 MiB
  float* h    = (float*)(ws + (4u << 20));       // 32 MiB  [M,Di] fp32
  u16*   hb   = (u16*)(ws + (36u << 20));        // 16 MiB  [M,Di] bf16

  // 1) weights -> bf16 (tiny, L2-resident afterwards)
  {
    int n4 = (Di * D) / 4;                       // 262144 each
    f32_to_bf16_v4<<<(n4 + 255) / 256, 256, 0, stream>>>(
        (const float4*)Wd, (u32*)Wd_b, n4);
    f32_to_bf16_v4<<<(n4 + 255) / 256, 256, 0, stream>>>(
        (const float4*)Wu, (u32*)Wu_b, n4);
  }

  // 2) down-projection: h[M,Di] = x[M,D] * Wd[Di,D]^T
  gemm_wmma_bf16<false><<<dim3(Di / 128, M / 128), 256, 0, stream>>>(
      (const void*)x, Wd_b, h, M, Di, D);

  // 3) fused 3-layer EMA scan, bf16 output
  ema_scan3<<<(Bb * Di) / 256, 256, 0, stream>>>(h, log_a, hb);

  // 4) up-projection: out[M,D] = hb[M,Di] * Wu[D,Di]^T
  gemm_wmma_bf16<true><<<dim3(D / 128, M / 128), 256, 0, stream>>>(
      (const void*)hb, Wu_b, out, M, D, Di);
}
